// NodePredictor_3083786518816
// MI455X (gfx1250) — matrix operand order repaired and checked by reading, not compiled, not run on hardware
//
#include <hip/hip_runtime.h>

// ---- CDNA5 (gfx1250) wave32 WMMA types ----
typedef __attribute__((ext_vector_type(16))) __bf16         v16bf;
typedef __attribute__((ext_vector_type(8)))  float          v8f;
typedef __attribute__((ext_vector_type(8)))  unsigned short us8;
typedef __attribute__((ext_vector_type(4)))  unsigned int   u32x4;

#define BM   128   // rows (batch) per workgroup
#define BN   128   // cols (DOUT) per workgroup
#define BK   32    // K step == WMMA K for bf16
#define LDK  40    // padded LDS row stride in bf16 elements (80 B, conflict-free b128)
#define TILE (BM * LDK)   // 5120 ushorts per plane (A and B tiles same size)

#define NNODE 64
#define DIN   512
#define DOUT  512
#define BATCH 2048
#define NK    (DIN / BK)

static __device__ __forceinline__ float bf16_to_f32(unsigned short h) {
    unsigned int u = ((unsigned int)h) << 16;
    return __builtin_bit_cast(float, u);
}
// split fp32 into bf16 hi + bf16 residual (bf16x3 GEMM trick, near-fp32 accuracy)
// uses the native f32->bf16 converter (RNE) instead of integer emulation: ~4 VALU/float
static __device__ __forceinline__ void split2(float v, unsigned short& hi, unsigned short& lo) {
    const __bf16 h = (__bf16)v;                       // v_cvt (pk) bf16, RNE
    const unsigned short hu = __builtin_bit_cast(unsigned short, h);
    const float r = v - bf16_to_f32(hu);              // exact residual (Sterbenz)
    const __bf16 l = (__bf16)r;
    hi = hu;
    lo = __builtin_bit_cast(unsigned short, l);
}

union FragU { u32x4 q[2]; v16bf v; };

// load 16 bf16 (two 16B chunks) from LDS into a WMMA fragment
static __device__ __forceinline__ v16bf load_frag(const unsigned short* p, int e0, int e1) {
    FragU f;
    f.q[0] = *(const u32x4*)(p + e0);
    f.q[1] = *(const u32x4*)(p + e1);
    return f.v;
}

__global__ __launch_bounds__(256) void node_linear_wmma(
    const float* __restrict__ x,    // [B, N, DIN]
    const float* __restrict__ W,    // [N, DIN, DOUT]
    const float* __restrict__ bias, // [N, DOUT]
    float* __restrict__ out)        // [B, N*DOUT]
{
    // double-buffered: [buf][plane][tile]; planes 0=Ah 1=Al 2=Bh 3=Bl  (80 KB total)
    __shared__ __align__(16) unsigned short smem[2][4][TILE];

    const int tid  = threadIdx.x;
    const int lane = tid & 31;
    const int wave = tid >> 5;
    const int wm   = wave & 3;    // 4 wave-rows  of 32 output rows
    const int wn   = wave >> 2;   // 2 wave-cols  of 64 output cols
    const int half = lane >> 4;   // lane half (K-block select per ISA layout)
    const int l16  = lane & 15;

    const int n  = blockIdx.z;          // node
    const int b0 = blockIdx.y * BM;     // batch-row base
    const int o0 = blockIdx.x * BN;     // DOUT base

    v8f acc[2][4];
    const v8f vzero = {0.f, 0.f, 0.f, 0.f, 0.f, 0.f, 0.f, 0.f};
#pragma unroll
    for (int ti = 0; ti < 2; ++ti)
#pragma unroll
        for (int tj = 0; tj < 4; ++tj) acc[ti][tj] = vzero;

    // ---- stage one K-tile (global -> bf16 hi/lo split -> LDS buf) ----
    auto stage = [&](int buf, int k0) {
        unsigned short* Ah = &smem[buf][0][0];
        unsigned short* Al = &smem[buf][1][0];
        unsigned short* Bh = &smem[buf][2][0];
        unsigned short* Bl = &smem[buf][3][0];
        const float* xb = x + ((size_t)b0 * NNODE + n) * DIN + k0;
        const float* wb = W + ((size_t)n * DIN + k0) * DOUT + o0;

        // A tile: x[b0..+127, n, k0..+31] -> Ah/Al[row][k], 16B LDS stores
#pragma unroll
        for (int it = 0; it < 2; ++it) {
            const int g   = it * 256 + tid;   // 512 groups of 8 K
            const int row = g >> 2;           // 0..127
            const int kg  = g & 3;            // 0..3 (8 K each)
            const float* p = xb + (size_t)row * (NNODE * DIN) + kg * 8;
            const float4 v0 = *(const float4*)(p);
            const float4 v1 = *(const float4*)(p + 4);
            const float f[8] = {v0.x, v0.y, v0.z, v0.w, v1.x, v1.y, v1.z, v1.w};
            us8 hv, lv;
#pragma unroll
            for (int j = 0; j < 8; ++j) {
                unsigned short h, l;
                split2(f[j], h, l);
                hv[j] = h; lv[j] = l;
            }
            *(us8*)(Ah + row * LDK + kg * 8) = hv;
            *(us8*)(Al + row * LDK + kg * 8) = lv;
        }
        // B tile (transposed): W[n, k0..+31, o0..+127] -> Bh/Bl[col][k].
        // Each thread loads two adjacent K rows so the transposed store is a
        // packed dword {bf16(k), bf16(k+1)} -> full-bank 4B ds stores.
#pragma unroll
        for (int it = 0; it < 2; ++it) {
            const int g  = it * 256 + tid;    // 512 groups
            const int kp = g >> 5;            // 0..15 (K pair)
            const int cg = g & 31;            // 0..31 (4 cols each)
            const int k  = kp * 2;
            const float* p = wb + (size_t)k * DOUT + cg * 4;
            const float4 u0 = *(const float4*)(p);
            const float4 u1 = *(const float4*)(p + DOUT);
            const float f0[4] = {u0.x, u0.y, u0.z, u0.w};
            const float f1[4] = {u1.x, u1.y, u1.z, u1.w};
#pragma unroll
            for (int j = 0; j < 4; ++j) {
                unsigned short h0, l0, h1, l1;
                split2(f0[j], h0, l0);
                split2(f1[j], h1, l1);
                const int c = cg * 4 + j;
                *(unsigned int*)(Bh + c * LDK + k) =
                    (unsigned int)h0 | ((unsigned int)h1 << 16);
                *(unsigned int*)(Bl + c * LDK + k) =
                    (unsigned int)l0 | ((unsigned int)l1 << 16);
            }
        }
    };

    // ---- 24 WMMAs on one staged K-tile ----
    auto compute = [&](int buf) {
        const unsigned short* Ah = &smem[buf][0][0];
        const unsigned short* Al = &smem[buf][1][0];
        const unsigned short* Bh = &smem[buf][2][0];
        const unsigned short* Bl = &smem[buf][3][0];

        // A fragments (dense 16-bit A 16x32: K = half*8+{0..7}, 16+half*8+{0..7})
        v16bf afh[2], afl[2];
#pragma unroll
        for (int ti = 0; ti < 2; ++ti) {
            const int ra = (wm * 32 + ti * 16 + l16) * LDK + half * 8;
            afh[ti] = load_frag(Ah, ra, ra + 16);
            afl[ti] = load_frag(Al, ra, ra + 16);
        }
        // B fragments (dense 16-bit B 32x16: K = half*8+{0..7}, 16+half*8+{0..7})
#pragma unroll
        for (int tj = 0; tj < 4; ++tj) {
            const int cb = (wn * 64 + tj * 16 + l16) * LDK + half * 8;
            const v16bf bfh = load_frag(Bh, cb, cb + 16);
            const v16bf bfl = load_frag(Bl, cb, cb + 16);
#pragma unroll
            for (int ti = 0; ti < 2; ++ti) {
                // bf16x3: hh + hl + lh, fp32 accumulate
                acc[ti][tj] = __builtin_amdgcn_wmma_f32_16x16x32_bf16(
                    false, afh[ti], false, bfh, (short)0, acc[ti][tj], false, false);
                acc[ti][tj] = __builtin_amdgcn_wmma_f32_16x16x32_bf16(
                    false, afh[ti], false, bfl, (short)0, acc[ti][tj], false, false);
                acc[ti][tj] = __builtin_amdgcn_wmma_f32_16x16x32_bf16(
                    false, afl[ti], false, bfh, (short)0, acc[ti][tj], false, false);
            }
        }
    };

    // ---- software pipeline: stage(k+1) overlaps WMMA(k); one barrier/iter ----
    stage(0, 0);
    __syncthreads();
#pragma unroll 1
    for (int kt = 0; kt < NK; ++kt) {
        const int cur = kt & 1;
        if (kt + 1 < NK) stage(cur ^ 1, (kt + 1) * BK);
        compute(cur);
        __syncthreads();
    }

    // ---- epilogue: bias add + store (C/D layout: M = half*8 + v, N = l16) ----
    float bv[4];
#pragma unroll
    for (int tj = 0; tj < 4; ++tj)
        bv[tj] = bias[(size_t)n * DOUT + o0 + wn * 64 + tj * 16 + l16];

#pragma unroll
    for (int ti = 0; ti < 2; ++ti) {
        const int grow0 = b0 + wm * 32 + ti * 16 + half * 8;
        float* orow = out + (size_t)grow0 * (NNODE * DOUT) + (size_t)n * DOUT;
#pragma unroll
        for (int tj = 0; tj < 4; ++tj) {
            const int gcol = o0 + wn * 64 + tj * 16 + l16;
#pragma unroll
            for (int v = 0; v < 8; ++v) {
                orow[(size_t)v * (NNODE * DOUT) + gcol] = acc[ti][tj][v] + bv[tj];
            }
        }
    }
}

extern "C" void kernel_launch(void* const* d_in, const int* in_sizes, int n_in,
                              void* d_out, int out_size, void* d_ws, size_t ws_size,
                              hipStream_t stream) {
    const float* x  = (const float*)d_in[0];   // [2048, 64, 512]
    const float* W  = (const float*)d_in[1];   // [64, 512, 512]
    const float* b  = (const float*)d_in[2];   // [64, 512]
    float* out      = (float*)d_out;           // [2048, 64*512]

    dim3 grid(DOUT / BN, BATCH / BM, NNODE);   // (4, 16, 64)
    dim3 block(256);                            // 8 waves (wave32)
    node_linear_wmma<<<grid, block, 0, stream>>>(x, W, b, out);
}
